// MLPcpp_sparse_37958920962348
// MI455X (gfx1250) — compile-verified, matrix operand order checked by reading, not run on hardware
//
#include <hip/hip_runtime.h>

typedef __attribute__((ext_vector_type(16))) _Float16 v16h;
typedef __attribute__((ext_vector_type(8)))  _Float16 v8h;
typedef __attribute__((ext_vector_type(4)))  _Float16 v4h;
typedef __attribute__((ext_vector_type(8)))  float    v8f;
typedef __attribute__((ext_vector_type(4)))  int      v4i;

union FragH { v16h v; v8h h[2]; };

#define BM 128
#define BN 128
#define BK 64
#define LDSS (BK + 8)       // padded LDS row stride in halves (144B, 16B-aligned, kills bank conflicts)
#define ABUF (BM * LDSS)    // halves per buffer
#define NTHREADS 128        // 4 wave32s

// ---- CDNA5 async global->LDS (ASYNCcnt-tracked) ----
__device__ __forceinline__ void async_copy_b128(const _Float16* gptr, _Float16* lptr) {
#if __has_builtin(__builtin_amdgcn_global_load_async_to_lds_b128)
  __builtin_amdgcn_global_load_async_to_lds_b128(
      (__attribute__((address_space(1))) v4i*)(const __attribute__((address_space(1))) void*)gptr,
      (__attribute__((address_space(3))) v4i*)(__attribute__((address_space(3))) void*)lptr,
      0, 0);
#else
  unsigned lds_off = (unsigned)(uintptr_t)(__attribute__((address_space(3))) void*)lptr;
  asm volatile("global_load_async_to_lds_b128 %0, %1, off"
               :: "v"(lds_off), "v"(gptr) : "memory");
#endif
}

__device__ __forceinline__ void wait_async0() {
#if __has_builtin(__builtin_amdgcn_s_wait_asynccnt)
  __builtin_amdgcn_s_wait_asynccnt(0);
#else
  asm volatile("s_wait_asynccnt 0" ::: "memory");
#endif
}

// ---------------- f32 -> f16 conversion (one pass over x) ----------------
__global__ __launch_bounds__(256) void cvt_f32_to_f16(const float* __restrict__ src,
                                                      _Float16* __restrict__ dst, int n) {
  int i = (blockIdx.x * 256 + threadIdx.x) * 4;
  if (i < n) {
    float4 f = *(const float4*)(src + i);
    v4h h;
    h[0] = (_Float16)f.x; h[1] = (_Float16)f.y;
    h[2] = (_Float16)f.z; h[3] = (_Float16)f.w;
    *(v4h*)(dst + i) = h;
  }
}

// ---------------- WMMA GEMM: C[M,N] = act(A[M,K] @ W[N,K]^T) ----------------
// Software-pipelined, double-buffered LDS:
//   - A (f16 activations): async b128 copies for tile k+1 issued right after the
//     barrier of tile k; they land during the 32 WMMAs of tile k (ASYNCcnt).
//   - W (f32 weights): f32->f16 converted in-register; loads for tile k+1 are
//     split in two halves wrapped around the two 16-WMMA compute halves so the
//     global latency is covered by WMMA issue, holding only 32 staging VGPRs.
template<bool RELU, bool HALF_OUT>
__global__ __launch_bounds__(NTHREADS) void mlp_gemm(const _Float16* __restrict__ A,
                                                     const float* __restrict__ W,
                                                     void* __restrict__ Cout,
                                                     int N, int K) {
  __shared__ _Float16 sA[2 * ABUF];
  __shared__ _Float16 sW[2 * ABUF];

  const int t    = threadIdx.x;
  const int lane = t & 31;
  const int wave = t >> 5;     // 0..3
  const int wm   = wave & 1;   // 2 wave-rows (64 M-rows each)
  const int wn   = wave >> 1;  // 2 wave-cols (64 N-cols each)

  const int blockM = blockIdx.y * BM;
  const int blockN = blockIdx.x * BN;

  // Staging: thread t owns the full 64-element K-slice of row t in both tiles
  const _Float16* gA = A + (size_t)(blockM + t) * K;
  const float*    gW = W + (size_t)(blockN + t) * K;

  v8f c[4][4] = {};
  float4 wreg[8];
  const int nk = K / BK;

  auto stage_A_async = [&](int buf, const _Float16* g) {
    _Float16* l = &sA[buf * ABUF + t * LDSS];
    #pragma unroll
    for (int i = 0; i < 8; i++)
      async_copy_b128(g + i * 8, l + i * 8);
  };
  auto load_W_half = [&](const float* g, int half) {
    #pragma unroll
    for (int i = 0; i < 4; i++) {
      wreg[2 * i]     = *(const float4*)(g + half * 32 + i * 8);
      wreg[2 * i + 1] = *(const float4*)(g + half * 32 + i * 8 + 4);
    }
  };
  auto store_W_half = [&](int buf, int half) {
    _Float16* l = &sW[buf * ABUF + t * LDSS + half * 32];
    #pragma unroll
    for (int i = 0; i < 4; i++) {
      float4 f0 = wreg[2 * i], f1 = wreg[2 * i + 1];
      v8h h;
      h[0] = (_Float16)f0.x; h[1] = (_Float16)f0.y;
      h[2] = (_Float16)f0.z; h[3] = (_Float16)f0.w;
      h[4] = (_Float16)f1.x; h[5] = (_Float16)f1.y;
      h[6] = (_Float16)f1.z; h[7] = (_Float16)f1.w;
      *(v8h*)(l + i * 8) = h;
    }
  };
  auto compute_half = [&](int buf, int ks) {
    // ISA 16-bit A/B fragment layout: lanes 0-15 take K {0..7,16..23},
    // lanes 16-31 take K {8..15,24..31} of the 32-deep step.
    const int klo = ks * 32 + ((lane >> 4) << 3);
    const int bo  = buf * ABUF;
    FragH a[4], b[4];
    #pragma unroll
    for (int im = 0; im < 4; im++) {
      const _Float16* p = &sA[bo + (wm * 64 + im * 16 + (lane & 15)) * LDSS + klo];
      a[im].h[0] = *(const v8h*)(p);
      a[im].h[1] = *(const v8h*)(p + 16);
    }
    #pragma unroll
    for (int jn = 0; jn < 4; jn++) {
      const _Float16* p = &sW[bo + (wn * 64 + jn * 16 + (lane & 15)) * LDSS + klo];
      b[jn].h[0] = *(const v8h*)(p);
      b[jn].h[1] = *(const v8h*)(p + 16);
    }
    #pragma unroll
    for (int im = 0; im < 4; im++)
      #pragma unroll
      for (int jn = 0; jn < 4; jn++)
        c[im][jn] = __builtin_amdgcn_wmma_f32_16x16x32_f16(
            false, a[im].v, false, b[jn].v, (short)0, c[im][jn], false, false);
  };

  // ---- prologue: stage tile 0 into buffer 0 ----
  stage_A_async(0, gA);
  load_W_half(gW, 0); store_W_half(0, 0);
  load_W_half(gW, 1); store_W_half(0, 1);

  // ---- pipelined main loop: one barrier per K-tile ----
  for (int k = 0; k < nk; k++) {
    const int cur = k & 1, nxt = cur ^ 1;
    const bool has_next = (k + 1 < nk);

    wait_async0();       // tile k's async A copies have landed
    __syncthreads();     // everyone's buffer-cur writes visible; buffer-nxt free

    if (has_next) {
      stage_A_async(nxt, gA + BK);            // overlaps all 32 WMMAs below
      load_W_half(gW + BK, 0);                // latency hidden by compute_half 0
      if (k + 2 < nk)
        __builtin_prefetch(gW + 2 * BK, 0, 3);
    }
    compute_half(cur, 0);                     // 16 x v_wmma
    if (has_next) {
      store_W_half(nxt, 0);
      load_W_half(gW + BK, 1);                // latency hidden by compute_half 1
    }
    compute_half(cur, 1);                     // 16 x v_wmma
    if (has_next) {
      store_W_half(nxt, 1);
      gA += BK; gW += BK;
    }
  }

  // ---- write back: C/D layout = VGPR v -> M=v (lanes 0-15) / M=v+8 (lanes 16-31) ----
  const int colBase = blockN + wn * 64;
  #pragma unroll
  for (int im = 0; im < 4; im++) {
    const int row0 = blockM + wm * 64 + im * 16 + ((lane >> 4) << 3);
    #pragma unroll
    for (int jn = 0; jn < 4; jn++) {
      const int col = colBase + jn * 16 + (lane & 15);
      #pragma unroll
      for (int v = 0; v < 8; v++) {
        float val = c[im][jn][v];
        if (RELU) val = fmaxf(val, 0.0f);
        const size_t idx = (size_t)(row0 + v) * N + col;
        if (HALF_OUT) ((_Float16*)Cout)[idx] = (_Float16)val;
        else          ((float*)Cout)[idx]    = val;
      }
    }
  }
}

extern "C" void kernel_launch(void* const* d_in, const int* in_sizes, int n_in,
                              void* d_out, int out_size, void* d_ws, size_t ws_size,
                              hipStream_t stream) {
  const float* x     = (const float*)d_in[0];  // [1024, 4096]
  const float* W_in  = (const float*)d_in[1];  // [4096, 4096]
  const float* W_hid = (const float*)d_in[2];  // [3, 4096, 4096]
  const float* W_out = (const float*)d_in[3];  // [1024, 4096]

  const int Bb = 1024, D = 4096, OUT = 1024;

  _Float16* act0 = (_Float16*)d_ws;             // 8 MB
  _Float16* act1 = act0 + (size_t)Bb * D;       // 8 MB

  cvt_f32_to_f16<<<(Bb * D / 4 + 255) / 256, 256, 0, stream>>>(x, act0, Bb * D);

  dim3 blk(NTHREADS);
  dim3 g1(D / BN, Bb / BM);   // 32 x 8 workgroups

  mlp_gemm<true, true><<<g1, blk, 0, stream>>>(act0, W_in, (void*)act1, D, D);

  for (int l = 0; l < 3; l++) {
    _Float16* src = (l & 1) ? act0 : act1;
    _Float16* dst = (l & 1) ? act1 : act0;
    mlp_gemm<true, true><<<g1, blk, 0, stream>>>(src, W_hid + (size_t)l * D * D,
                                                 (void*)dst, D, D);
  }
  // l=0: act1->act0, l=1: act0->act1, l=2: act1->act0 => final activations in act0

  dim3 g2(OUT / BN, Bb / BM);
  mlp_gemm<false, false><<<g2, blk, 0, stream>>>(act0, W_out, d_out, OUT, D);
}
